// RefinerBlock_53781580480572
// MI455X (gfx1250) — matrix operand order repaired and checked by reading, not compiled, not run on hardware
//
#include <hip/hip_runtime.h>
#include <math.h>

// ---------------------------------------------------------------------------
// MI455X (gfx1250) transformer block: RMSNorm -> QKV -> causal flash-attn ->
// O-proj(+res) -> RMSNorm -> SwiGLU MLP(+res).  All big contractions run on
// v_wmma_f32_16x16x32_bf16 (fp32 accumulate).  GEMM staging uses the CDNA5
// async copy path: global_load_async_to_lds_b128 + s_wait_asynccnt, double
// buffered, last tile peeled so the steady-state loop is branch-free.
// B=2,S=2048,D=1024,H=16,HD=64,FF=4096 hardcoded.
// ---------------------------------------------------------------------------

typedef unsigned int  u32;
typedef unsigned short u16;
typedef unsigned long long u64;
typedef __attribute__((ext_vector_type(16))) __bf16 v16bf;
typedef __attribute__((ext_vector_type(8)))  float  v8f;

union AFrag { v16bf f; u32 u[8]; };

__device__ __forceinline__ u16 f2bf(float x) {
  union { float f; u32 u; } v; v.f = x;
  u32 r = v.u + 0x7FFFu + ((v.u >> 16) & 1u);   // round-to-nearest-even
  return (u16)(r >> 16);
}
__device__ __forceinline__ float bf2f(u16 h) {
  union { u32 u; float f; } v; v.u = ((u32)h) << 16; return v.f;
}
__device__ __forceinline__ v8f zero8() {
  v8f z = {0.f,0.f,0.f,0.f,0.f,0.f,0.f,0.f}; return z;
}

// CDNA5 async copy: 16B global -> LDS, tracked by ASYNCcnt (no VGPR transit).
// lds_off: wave-relative LDS byte address (low 32 bits of the flat pointer,
// which the ISA takes as the LDS address).
__device__ __forceinline__ void async_cp16(u32 lds_off, const void* gptr) {
  asm volatile("global_load_async_to_lds_b128 %0, %1, off"
               :: "v"(lds_off), "v"((u64)(size_t)gptr) : "memory");
}

#define NEGINF (-__builtin_inff())

// ---------------------------------------------------------------------------
// Transpose + f32->bf16 convert:  in[Kr,Nc] f32  ->  out[Nc,Kr] bf16
// ---------------------------------------------------------------------------
__global__ __launch_bounds__(256) void transcvt_k(const float* __restrict__ in,
                                                  u16* __restrict__ out,
                                                  int Kr, int Nc) {
  __shared__ float t[32][33];
  const int kb = blockIdx.y * 32, nb = blockIdx.x * 32;
  const int tx = threadIdx.x & 31, ty = threadIdx.x >> 5;
#pragma unroll
  for (int j = 0; j < 4; j++)
    t[ty + j * 8][tx] = in[(size_t)(kb + ty + j * 8) * Nc + nb + tx];
  __syncthreads();
#pragma unroll
  for (int j = 0; j < 4; j++)
    out[(size_t)(nb + ty + j * 8) * Kr + kb + tx] = f2bf(t[tx][ty + j * 8]);
}

// ---------------------------------------------------------------------------
// RMSNorm: x[row,1024] f32 -> out[row,1024] bf16   (one block per row)
// ---------------------------------------------------------------------------
__global__ __launch_bounds__(256) void rmsnorm_k(const float* __restrict__ x,
                                                 const float* __restrict__ w,
                                                 u16* __restrict__ out) {
  __shared__ float red[8];
  const int row = blockIdx.x, tid = threadIdx.x;
  const int lane = tid & 31, wave = tid >> 5;
  const float* xr = x + (size_t)row * 1024;
  float v0 = xr[tid], v1 = xr[tid + 256], v2 = xr[tid + 512], v3 = xr[tid + 768];
  float ss = v0 * v0 + v1 * v1 + v2 * v2 + v3 * v3;
  ss += __shfl_xor(ss, 16, 32);
  ss += __shfl_xor(ss, 8, 32);
  ss += __shfl_xor(ss, 4, 32);
  ss += __shfl_xor(ss, 2, 32);
  ss += __shfl_xor(ss, 1, 32);
  if (lane == 0) red[wave] = ss;
  __syncthreads();
  float tot = 0.f;
#pragma unroll
  for (int i = 0; i < 8; i++) tot += red[i];
  const float r = rsqrtf(tot * (1.0f / 1024.0f) + 1e-6f);
  u16* orow = out + (size_t)row * 1024;
  orow[tid]       = f2bf(w[tid] * v0 * r);
  orow[tid + 256] = f2bf(w[tid + 256] * v1 * r);
  orow[tid + 512] = f2bf(w[tid + 512] * v2 * r);
  orow[tid + 768] = f2bf(w[tid + 768] * v3 * r);
}

// ---------------------------------------------------------------------------
// bf16 WMMA GEMM:  C[M,N] = A[M,K] * BT[N,K]^T (+bias) (+residual)
// Block tile 64x128, BK=32.  8 waves (2x4), each wave a 32x32 tile via
// 2x2 v_wmma_f32_16x16x32_bf16.
// LDS: double-buffered, row stride 72 halfwords (144B: 16B-aligned async
// stores; r*36 mod 64 is a permutation over 16 lanes -> conflict-free reads).
// Staging: global_load_async_to_lds_b128 (ASYNCcnt); steady-state loop is
// branch-free (unconditional prefetch of tile t+1, s_wait_asynccnt 0x3);
// last tile peeled into an epilogue with s_wait_asynccnt 0x0.
// store mode: 0=bf16[M,N], 1=bf16 [B,H,S,HD] (q/k), 2=bf16 [B,H,HD,S] (v),
//             3=f32[M,N] + residual
// ---------------------------------------------------------------------------
#define LDW 72          // LDS row stride, halfwords
#define LDWU 36         // LDS row stride, dwords

__global__ __launch_bounds__(256) void gemm_k(const u16* __restrict__ A,
                                              const u16* __restrict__ BT,
                                              const float* __restrict__ bias,
                                              void* __restrict__ out,
                                              const float* __restrict__ resid,
                                              int M, int N, int Kd, int mode) {
  __shared__ __align__(16) u16 sA[2][64 * LDW];
  __shared__ __align__(16) u16 sB[2][128 * LDW];
  const int tid = threadIdx.x, lane = tid & 31, wave = tid >> 5;
  const int l15 = lane & 15, hf = lane >> 4;
  const int wm = wave >> 2, wn = wave & 3;
  const int m0 = blockIdx.y * 64, n0 = blockIdx.x * 128;

  v8f acc[2][2];
#pragma unroll
  for (int i = 0; i < 2; i++)
#pragma unroll
    for (int j = 0; j < 2; j++) acc[i][j] = zero8();

  // ---- per-thread staging slots (fixed): 8 bf16 (16B) per async op ----
  const int rowA = tid >> 2, segA = tid & 3;               // 64 rows x 4 segs
  const int rowB0 = tid >> 2, rowB1 = 64 + (tid >> 2);     // 128 rows x 4 segs
  const int segB = tid & 3;
  const u16* gA  = A  + (size_t)(m0 + rowA)  * Kd + segA * 8;
  const u16* gB0 = BT + (size_t)(n0 + rowB0) * Kd + segB * 8;
  const u16* gB1 = BT + (size_t)(n0 + rowB1) * Kd + segB * 8;
  u32 ldsA[2], ldsB0[2], ldsB1[2];
#pragma unroll
  for (int b = 0; b < 2; b++) {
    ldsA[b]  = (u32)(size_t)(const void*)&sA[b][rowA * LDW + segA * 8];
    ldsB0[b] = (u32)(size_t)(const void*)&sB[b][rowB0 * LDW + segB * 8];
    ldsB1[b] = (u32)(size_t)(const void*)&sB[b][rowB1 * LDW + segB * 8];
  }

  // per-wave fragment load + 4 WMMAs from one LDS buffer
  auto compute_tile = [&](int cur) {
    const u32* sAu = (const u32*)sA[cur];
    const u32* sBu = (const u32*)sB[cur];
    AFrag af[2], bfr[2];
#pragma unroll
    for (int sm = 0; sm < 2; sm++) {        // A frag: row per lane
      const int r = wm * 32 + sm * 16 + l15;
      const int base = r * LDWU + hf * 8;
#pragma unroll
      for (int v = 0; v < 4; v++) {
        af[sm].u[v]     = sAu[base + v];
        af[sm].u[4 + v] = sAu[base + 4 + v];
      }
    }
#pragma unroll
    for (int sn = 0; sn < 2; sn++) {        // B frag: K-pairs per VGPR
      const int n = wn * 32 + sn * 16 + l15;
      const int base = n * LDWU + hf * 8;
#pragma unroll
      for (int v = 0; v < 8; v++) bfr[sn].u[v] = sBu[base + v];
    }
#pragma unroll
    for (int sm = 0; sm < 2; sm++)
#pragma unroll
      for (int sn = 0; sn < 2; sn++)
        acc[sm][sn] = __builtin_amdgcn_wmma_f32_16x16x32_bf16(
            false, af[sm].f, false, bfr[sn].f, (short)0, acc[sm][sn],
            false, false);
  };

  // prefetch tile 0
  async_cp16(ldsA[0], gA);
  async_cp16(ldsB0[0], gB0);
  async_cp16(ldsB1[0], gB1);

  const int nt = Kd >> 5;                   // always >= 2 here
  for (int t = 0; t < nt - 1; ++t) {        // branch-free steady state
    const int cur = t & 1;
    const size_t ko = (size_t)(t + 1) * 32;
    async_cp16(ldsA[cur ^ 1],  gA + ko);    // prefetch next tile -> other buf
    async_cp16(ldsB0[cur ^ 1], gB0 + ko);
    async_cp16(ldsB1[cur ^ 1], gB1 + ko);
    asm volatile("s_wait_asynccnt 0x3" ::: "memory");  // tile t complete
    __syncthreads();                        // all waves' portions visible
    compute_tile(cur);
    __syncthreads();                        // reads done before buf reuse
  }
  // epilogue: last tile
  asm volatile("s_wait_asynccnt 0x0" ::: "memory");
  __syncthreads();
  compute_tile((nt - 1) & 1);

#pragma unroll
  for (int sm = 0; sm < 2; sm++) {
#pragma unroll
    for (int sn = 0; sn < 2; sn++) {
      const int col = n0 + wn * 32 + sn * 16 + l15;
      const float bb = bias ? bias[col] : 0.0f;
#pragma unroll
      for (int v = 0; v < 8; v++) {
        const int row = m0 + wm * 32 + sm * 16 + hf * 8 + v;
        const float val = acc[sm][sn][v] + bb;
        if (mode == 0) {
          ((u16*)out)[(size_t)row * N + col] = f2bf(val);
        } else if (mode == 1) {   // -> [B,H,S,HD]
          const int b = row >> 11, s = row & 2047, h = col >> 6, hd = col & 63;
          ((u16*)out)[((((size_t)b * 16 + h) * 2048 + s) << 6) + hd] = f2bf(val);
        } else if (mode == 2) {   // -> [B,H,HD,S]
          const int b = row >> 11, s = row & 2047, h = col >> 6, hd = col & 63;
          ((u16*)out)[((((size_t)b * 16 + h) * 64 + hd) << 11) + s] = f2bf(val);
        } else {                  // f32 + residual
          ((float*)out)[(size_t)row * N + col] =
              val + resid[(size_t)row * N + col];
        }
      }
    }
  }
}

// ---------------------------------------------------------------------------
// Causal flash attention.  grid = (S/128, B*H), block = 256 (8 waves).
// Each wave owns 16 query rows; streams 32-key tiles to its causal limit.
// Q [B,H,S,HD] bf16, K [B,H,S,HD] bf16, VT [B,H,HD,S] bf16,
// Out [B,S,D] bf16.  Per-wave LDS slab re-lays P from C-layout to A-layout
// (ordered with s_wait_dscnt; no block barrier -> no deadlock on unequal
// causal trip counts).
// ---------------------------------------------------------------------------
__global__ __launch_bounds__(256) void attn_k(const u16* __restrict__ Q,
                                              const u16* __restrict__ K,
                                              const u16* __restrict__ VT,
                                              const int* __restrict__ amask,
                                              u16* __restrict__ Out) {
  __shared__ __align__(16) u16 sP[8 * 16 * 32];
  const int tid = threadIdx.x, lane = tid & 31, wave = tid >> 5;
  const int l15 = lane & 15, hf = lane >> 4;
  const int bh = blockIdx.y, b = bh >> 4, h = bh & 15;
  const int qbase = blockIdx.x * 128 + wave * 16;

  // Q A-fragments for both HD halves (contraction chunks of 32)
  const u32* qr = (const u32*)(Q + (((size_t)bh * 2048) + qbase + l15) * 64);
  AFrag aq0, aq1;
#pragma unroll
  for (int v = 0; v < 4; v++) {
    aq0.u[v]     = qr[hf * 8 + v];
    aq0.u[4 + v] = qr[4 + hf * 8 + v];
    aq1.u[v]     = qr[16 + hf * 8 + v];
    aq1.u[4 + v] = qr[20 + hf * 8 + v];
  }

  v8f oacc[4];
#pragma unroll
  for (int j = 0; j < 4; j++) oacc[j] = zero8();
  float mi[8], li[8];
#pragma unroll
  for (int v = 0; v < 8; v++) { mi[v] = NEGINF; li[v] = 0.0f; }

  u16* myP  = sP + wave * 512;
  u32* myPu = (u32*)myP;
  const int qmax = qbase + 15;
  const int* am = amask + (size_t)b * 2048;
  const u32* kbase = (const u32*)(K + ((size_t)bh * 2048) * 64);
  const u32* vbase[4];
#pragma unroll
  for (int j = 0; j < 4; j++)
    vbase[j] = (const u32*)(VT + (((size_t)bh * 64) + j * 16 + l15) * 2048);

  for (int k0 = 0; k0 <= qmax; k0 += 32) {
    // ---- scores: 16x32 keys, contraction over HD=64 -> 4 WMMAs ----
    v8f sc0 = zero8(), sc1 = zero8();
    {
      const u32* kr0 = kbase + (size_t)(k0 + l15) * 32;
      const u32* kr1 = kbase + (size_t)(k0 + 16 + l15) * 32;
      AFrag bk;
#pragma unroll
      for (int v = 0; v < 8; v++) bk.u[v] = kr0[hf * 8 + v];
      sc0 = __builtin_amdgcn_wmma_f32_16x16x32_bf16(false, aq0.f, false, bk.f,
                                                    (short)0, sc0, false, false);
#pragma unroll
      for (int v = 0; v < 8; v++) bk.u[v] = kr0[16 + hf * 8 + v];
      sc0 = __builtin_amdgcn_wmma_f32_16x16x32_bf16(false, aq1.f, false, bk.f,
                                                    (short)0, sc0, false, false);
#pragma unroll
      for (int v = 0; v < 8; v++) bk.u[v] = kr1[hf * 8 + v];
      sc1 = __builtin_amdgcn_wmma_f32_16x16x32_bf16(false, aq0.f, false, bk.f,
                                                    (short)0, sc1, false, false);
#pragma unroll
      for (int v = 0; v < 8; v++) bk.u[v] = kr1[16 + hf * 8 + v];
      sc1 = __builtin_amdgcn_wmma_f32_16x16x32_bf16(false, aq1.f, false, bk.f,
                                                    (short)0, sc1, false, false);
    }
    const int key0 = k0 + l15, key1 = k0 + 16 + l15;
    const int am0 = am[key0], am1 = am[key1];

    float alpha_a[8];
#pragma unroll
    for (int v = 0; v < 8; v++) {
      const int row = qbase + hf * 8 + v;
      float s0 = sc0[v] * 0.125f;   // 1/sqrt(64)
      float s1 = sc1[v] * 0.125f;
      if (key0 > row || am0 == 0) s0 = NEGINF;
      if (key1 > row || am1 == 0) s1 = NEGINF;
      float mv = fmaxf(s0, s1);     // row max across the 16 lanes of a half
      mv = fmaxf(mv, __shfl_xor(mv, 1, 32));
      mv = fmaxf(mv, __shfl_xor(mv, 2, 32));
      mv = fmaxf(mv, __shfl_xor(mv, 4, 32));
      mv = fmaxf(mv, __shfl_xor(mv, 8, 32));
      float nm = fmaxf(mi[v], mv);
      float alpha, p0, p1;
      if (nm <= -1e30f) {
        alpha = 1.0f; p0 = 0.0f; p1 = 0.0f;
      } else {
        alpha = (mi[v] <= -1e30f) ? 0.0f : __expf(mi[v] - nm);
        p0 = (s0 <= -1e30f) ? 0.0f : __expf(s0 - nm);
        p1 = (s1 <= -1e30f) ? 0.0f : __expf(s1 - nm);
      }
      float ps = p0 + p1;
      ps += __shfl_xor(ps, 1, 32);
      ps += __shfl_xor(ps, 2, 32);
      ps += __shfl_xor(ps, 4, 32);
      ps += __shfl_xor(ps, 8, 32);
      li[v] = li[v] * alpha + ps;
      mi[v] = nm;
      alpha_a[v] = alpha;
      myP[(hf * 8 + v) * 32 + l15]      = f2bf(p0);   // C-layout -> LDS
      myP[(hf * 8 + v) * 32 + 16 + l15] = f2bf(p1);
    }
#pragma unroll
    for (int j = 0; j < 4; j++)
#pragma unroll
      for (int v = 0; v < 8; v++) oacc[j][v] *= alpha_a[v];

    // wave-local ordering of LDS stores before A-fragment reload
    asm volatile("s_wait_dscnt 0" ::: "memory");

    AFrag ap;                                   // P as 16x32 A-fragment
    {
      const int base = l15 * 16 + hf * 8;
#pragma unroll
      for (int v = 0; v < 4; v++) {
        ap.u[v]     = myPu[base + v];
        ap.u[4 + v] = myPu[base + 4 + v];
      }
    }
#pragma unroll
    for (int j = 0; j < 4; j++) {               // P·V -> 4 WMMAs
      AFrag bv;
#pragma unroll
      for (int v = 0; v < 8; v++) bv.u[v] = vbase[j][(k0 >> 1) + hf * 8 + v];
      oacc[j] = __builtin_amdgcn_wmma_f32_16x16x32_bf16(
          false, ap.f, false, bv.f, (short)0, oacc[j], false, false);
    }
  }

#pragma unroll
  for (int v = 0; v < 8; v++) {
    const int row = qbase + hf * 8 + v;
    const float inv = (li[v] > 0.0f) ? 1.0f / li[v] : 0.0f;
#pragma unroll
    for (int j = 0; j < 4; j++) {
      Out[(((size_t)b * 2048 + row) << 10) + h * 64 + j * 16 + l15] =
          f2bf(oacc[j][v] * inv);
    }
  }
}

// ---------------------------------------------------------------------------
// elementwise: gu = bf16( silu(gate) * up )
// ---------------------------------------------------------------------------
__global__ __launch_bounds__(256) void silumul_k(const u16* __restrict__ g,
                                                 const u16* __restrict__ u,
                                                 u16* __restrict__ o) {
  const size_t base = ((size_t)blockIdx.x * 256 + threadIdx.x) * 4;
#pragma unroll
  for (int i = 0; i < 4; i++) {
    const float gv = bf2f(g[base + i]);
    const float uv = bf2f(u[base + i]);
    const float s = gv / (1.0f + __expf(-gv));
    o[base + i] = f2bf(s * uv);
  }
}

// ---------------------------------------------------------------------------
extern "C" void kernel_launch(void* const* d_in, const int* in_sizes, int n_in,
                              void* d_out, int out_size, void* d_ws,
                              size_t ws_size, hipStream_t stream) {
  (void)in_sizes; (void)n_in; (void)out_size; (void)ws_size;
  const float* hidden  = (const float*)d_in[0];
  const int*   amask   = (const int*)d_in[1];
  const float* w_norm1 = (const float*)d_in[2];
  const float* wq = (const float*)d_in[3];
  const float* bq = (const float*)d_in[4];
  const float* wk = (const float*)d_in[5];
  const float* bk = (const float*)d_in[6];
  const float* wv = (const float*)d_in[7];
  const float* bv = (const float*)d_in[8];
  const float* wo = (const float*)d_in[9];
  const float* bo = (const float*)d_in[10];
  const float* w_norm2 = (const float*)d_in[11];
  const float* w_gate  = (const float*)d_in[12];
  const float* w_up    = (const float*)d_in[13];
  const float* w_down  = (const float*)d_in[14];

  const size_t MT = 4096;   // B*S tokens
  const size_t D  = 1024;
  const size_t FF = 4096;

  char* p = (char*)d_ws;
  auto alloc = [&](size_t bytes) -> char* {
    char* r = p;
    p += (bytes + 255) & ~(size_t)255;
    return r;
  };
  u16* wqT   = (u16*)alloc(D * D * 2);
  u16* wkT   = (u16*)alloc(D * D * 2);
  u16* wvT   = (u16*)alloc(D * D * 2);
  u16* woT   = (u16*)alloc(D * D * 2);
  u16* wgT   = (u16*)alloc(D * FF * 2);
  u16* wuT   = (u16*)alloc(D * FF * 2);
  u16* wdT   = (u16*)alloc(FF * D * 2);
  u16* h1b   = (u16*)alloc(MT * D * 2);
  u16* qb    = (u16*)alloc(MT * D * 2);      // [B,H,S,HD]
  u16* kb    = (u16*)alloc(MT * D * 2);      // [B,H,S,HD]
  u16* vtb   = (u16*)alloc(MT * D * 2);      // [B,H,HD,S]
  u16* attnb = (u16*)alloc(MT * D * 2);      // [B,S,D]
  float* x1  = (float*)alloc(MT * D * 4);
  u16* h2b   = (u16*)alloc(MT * D * 2);
  u16* gateb = (u16*)alloc(MT * FF * 2);
  u16* upb   = (u16*)alloc(MT * FF * 2);
  u16* gub   = (u16*)alloc(MT * FF * 2);

  // weight transpose+convert (weights then live bf16 in the 192MB L2)
  transcvt_k<<<dim3(32, 32), 256, 0, stream>>>(wq, wqT, 1024, 1024);
  transcvt_k<<<dim3(32, 32), 256, 0, stream>>>(wk, wkT, 1024, 1024);
  transcvt_k<<<dim3(32, 32), 256, 0, stream>>>(wv, wvT, 1024, 1024);
  transcvt_k<<<dim3(32, 32), 256, 0, stream>>>(wo, woT, 1024, 1024);
  transcvt_k<<<dim3(128, 32), 256, 0, stream>>>(w_gate, wgT, 1024, 4096);
  transcvt_k<<<dim3(128, 32), 256, 0, stream>>>(w_up, wuT, 1024, 4096);
  transcvt_k<<<dim3(32, 128), 256, 0, stream>>>(w_down, wdT, 4096, 1024);

  // attention sub-block
  rmsnorm_k<<<4096, 256, 0, stream>>>(hidden, w_norm1, h1b);
  gemm_k<<<dim3(8, 64), 256, 0, stream>>>(h1b, wqT, bq, qb, nullptr,
                                          4096, 1024, 1024, 1);
  gemm_k<<<dim3(8, 64), 256, 0, stream>>>(h1b, wkT, bk, kb, nullptr,
                                          4096, 1024, 1024, 1);
  gemm_k<<<dim3(8, 64), 256, 0, stream>>>(h1b, wvT, bv, vtb, nullptr,
                                          4096, 1024, 1024, 2);
  attn_k<<<dim3(16, 32), 256, 0, stream>>>(qb, kb, vtb, amask, attnb);
  gemm_k<<<dim3(8, 64), 256, 0, stream>>>(attnb, woT, bo, x1, hidden,
                                          4096, 1024, 1024, 3);

  // MLP sub-block
  rmsnorm_k<<<4096, 256, 0, stream>>>(x1, w_norm2, h2b);
  gemm_k<<<dim3(32, 64), 256, 0, stream>>>(h2b, wgT, nullptr, gateb, nullptr,
                                           4096, 4096, 1024, 0);
  gemm_k<<<dim3(32, 64), 256, 0, stream>>>(h2b, wuT, nullptr, upb, nullptr,
                                           4096, 4096, 1024, 0);
  silumul_k<<<16384, 256, 0, stream>>>(gateb, upb, gub);
  gemm_k<<<dim3(8, 64), 256, 0, stream>>>(gub, wdT, nullptr, d_out, x1,
                                          4096, 1024, 4096, 3);
}
